// Qwen2MoeSparseMoeBlock_65429531788003
// MI455X (gfx1250) — matrix operand order repaired and checked by reading, not compiled, not run on hardware
//
#include <hip/hip_runtime.h>
#include <hip/hip_bf16.h>
#include <math.h>

// ---------------------------------------------------------------------------
// Qwen2-MoE sparse block for gfx1250 (MI455X): bf16 WMMA GEMMs, fp32 routing,
// double-buffered LDS software pipeline.
// ---------------------------------------------------------------------------

typedef __bf16 bf16_t;
typedef bf16_t v16bf __attribute__((ext_vector_type(16)));
typedef bf16_t v8bf  __attribute__((ext_vector_type(8)));
typedef bf16_t v4bf  __attribute__((ext_vector_type(4)));
typedef float  v8f   __attribute__((ext_vector_type(8)));

constexpr int T    = 1024;   // tokens (B*S)
constexpr int H    = 2048;   // hidden
constexpr int E    = 16;     // experts
constexpr int TOPK = 4;
constexpr int IM   = 1408;   // moe intermediate
constexpr int ISH  = 5632;   // shared intermediate
constexpr int CAP  = 1024;   // per-expert assignment capacity (max = T)
constexpr int MT   = 128;    // rows per block tile (8 waves x 16)
constexpr int KST  = 32;     // k step (wmma 16x16x32 bf16)
constexpr int LDP  = 40;     // padded LDS row length in bf16 (80B, bank-friendly)

// ---- WMMA fragment helpers (layouts per CDNA5 ISA 7.12.2, wave32) ----------
// A 16x32 bf16: lane l -> row M = l&15; half = l>>4;
//   elems 0..7  = K = half*8 + j ; elems 8..15 = K = 16 + half*8 + j
__device__ inline v16bf frag_a(const bf16_t* row, int half) {
  v8bf lo = *(const v8bf*)(row + half * 8);
  v8bf hi = *(const v8bf*)(row + 16 + half * 8);
  v16bf r;
#pragma unroll
  for (int j = 0; j < 8; ++j) { r[j] = lo[j]; r[j + 8] = hi[j]; }
  return r;
}
// B 32x16 bf16: lane l -> col N = l&15; half = l>>4;
//   elems 0..7  = K = half*8 + j ; elems 8..15 = K = 16 + half*8 + j
__device__ inline v16bf frag_b(const bf16_t* row, int half) {
  v8bf lo = *(const v8bf*)(row + half * 8);
  v8bf hi = *(const v8bf*)(row + 16 + half * 8);
  v16bf r;
#pragma unroll
  for (int j = 0; j < 8; ++j) { r[j] = lo[j]; r[j + 8] = hi[j]; }
  return r;
}

__device__ inline v8f wmma_bf16(v16bf a, v16bf b, v8f c) {
  // (neg_a, A, neg_b, B, c_mod, C, reuse_a, reuse_b)
  return __builtin_amdgcn_wmma_f32_16x16x32_bf16(false, a, false, b,
                                                 (short)0, c, false, false);
}

__device__ inline float silu_f(float x) {
  return x * __builtin_amdgcn_rcpf(1.f + __expf(-x));     // v_rcp_f32, no div chain
}
__device__ inline float sigmoid_f(float x) {
  return __builtin_amdgcn_rcpf(1.f + __expf(-x));
}

#define ZERO8 {0.f,0.f,0.f,0.f,0.f,0.f,0.f,0.f}

// ---------------------------------------------------------------------------
// Router: fp32 logits (exact expert selection) + shared-expert gate logit.
// ---------------------------------------------------------------------------
__global__ void k_router(const float* __restrict__ x, const float* __restrict__ wr,
                         const float* __restrict__ wshg, float* __restrict__ logits,
                         float* __restrict__ sgate) {
  __shared__ float red[256];
  const int t = blockIdx.x, tid = threadIdx.x;
  const float* xr = x + (size_t)t * H;
  for (int e = 0; e < E + 1; ++e) {
    const float* w = (e < E) ? (wr + (size_t)e * H) : wshg;
    float s = 0.f;
    for (int h = tid; h < H; h += 256) s += xr[h] * w[h];
    red[tid] = s; __syncthreads();
    for (int off = 128; off > 0; off >>= 1) {
      if (tid < off) red[tid] += red[tid + off];
      __syncthreads();
    }
    if (tid == 0) { if (e < E) logits[(size_t)t * E + e] = red[0]; else sgate[t] = red[0]; }
    __syncthreads();
  }
}

__global__ void k_init(int* counts) { if (threadIdx.x < E) counts[threadIdx.x] = 0; }

// Softmax + top-4 per token; build per-expert gathered row lists.
__global__ void k_topk(const float* __restrict__ logits, int* __restrict__ counts,
                       int* __restrict__ rowmap, float* __restrict__ wmap) {
  const int t = blockIdx.x * blockDim.x + threadIdx.x;
  if (t >= T) return;
  float p[E]; float m = -1e30f;
#pragma unroll
  for (int e = 0; e < E; ++e) { p[e] = logits[(size_t)t * E + e]; m = fmaxf(m, p[e]); }
  float s = 0.f;
#pragma unroll
  for (int e = 0; e < E; ++e) { p[e] = __expf(p[e] - m); s += p[e]; }
  const float inv = 1.f / s;
#pragma unroll
  for (int e = 0; e < E; ++e) p[e] *= inv;
#pragma unroll
  for (int k = 0; k < TOPK; ++k) {
    int bj = 0; float bv = p[0];
#pragma unroll
    for (int e = 1; e < E; ++e) if (p[e] > bv) { bv = p[e]; bj = e; }
    p[bj] = -1.f;
    const int pos = atomicAdd(counts + bj, 1);
    rowmap[bj * CAP + pos] = t;
    wmap[bj * CAP + pos]   = bv;
  }
}

__global__ void k_cvtx(const float* __restrict__ x, bf16_t* __restrict__ xbf) {
  const int i = (blockIdx.x * 256 + threadIdx.x) * 4;    // grid == T*H/1024 exactly
  const float4 v = *(const float4*)(x + i);
  v4bf o; o[0] = (bf16_t)v.x; o[1] = (bf16_t)v.y; o[2] = (bf16_t)v.z; o[3] = (bf16_t)v.w;
  *(v4bf*)(xbf + i) = o;
}

// ---------------------------------------------------------------------------
// Expert gate/up: h = silu(x@Wg^T) * (x@Wu^T) on gathered rows.
// 128x16 tile, pipelined double-buffered LDS, 2 WMMAs per k-step per wave.
// ---------------------------------------------------------------------------
__global__ void k_moe_gateup(const bf16_t* __restrict__ xbf, const float* __restrict__ wg,
                             const float* __restrict__ wu, const int* __restrict__ counts,
                             const int* __restrict__ rowmap, bf16_t* __restrict__ hbuf) {
  __shared__ bf16_t As[2][MT][LDP];
  __shared__ bf16_t Bg[2][16][LDP];
  __shared__ bf16_t Bu[2][16][LDP];
  __shared__ int    toks[MT];
  const int e = blockIdx.y >> 3;            // CAP/MT == 8 tiles per expert
  const int mbase = (blockIdx.y & 7) * MT;
  const int ne = counts[e];
  if (mbase >= ne) return;
  const int n0 = blockIdx.x * 16;
  const int tid = threadIdx.x;
  if (tid < MT) {
    const int r = mbase + tid;
    toks[tid] = (r < ne) ? rowmap[e * CAP + r] : rowmap[e * CAP];  // pad w/ dup token
  }
  __syncthreads();
  const int lane = tid & 31, wave = tid >> 5;
  const int half = lane >> 4, nl = lane & 15;
  // staging geometry: A = 512 16B-chunks (2/thread), B = 2 floats/thread/matrix
  const int arow0 = tid >> 2, aoff = (tid & 3) * 8;
  const int brow = tid >> 4, boff = (tid & 15) * 2;
  const bf16_t* pA0 = xbf + (size_t)toks[arow0] * H + aoff;
  const bf16_t* pA1 = xbf + (size_t)toks[arow0 + 64] * H + aoff;
  const float*  pBg = wg + ((size_t)e * IM + n0 + brow) * H + boff;
  const float*  pBu = wu + ((size_t)e * IM + n0 + brow) * H + boff;
  v8f cg = ZERO8, cu = ZERO8;
  v8bf ra0, ra1; float rg0, rg1, ru0, ru1;
  auto gload = [&](int k0) {
    ra0 = *(const v8bf*)(pA0 + k0);
    ra1 = *(const v8bf*)(pA1 + k0);
    rg0 = pBg[k0]; rg1 = pBg[k0 + 1];
    ru0 = pBu[k0]; ru1 = pBu[k0 + 1];
  };
  auto lstore = [&](int b) {
    *(v8bf*)(&As[b][arow0][aoff]) = ra0;
    *(v8bf*)(&As[b][arow0 + 64][aoff]) = ra1;
    Bg[b][brow][boff] = (bf16_t)rg0; Bg[b][brow][boff + 1] = (bf16_t)rg1;
    Bu[b][brow][boff] = (bf16_t)ru0; Bu[b][brow][boff + 1] = (bf16_t)ru1;
  };
  gload(0); lstore(0);
  __syncthreads();
  constexpr int NK = H / KST;
  for (int ki = 0; ki < NK; ++ki) {
    const int cur = ki & 1;
    if (ki + 1 < NK) gload((ki + 1) * KST);       // in flight during WMMAs
    const v16bf a  = frag_a(&As[cur][wave * 16 + nl][0], half);
    const v16bf bg = frag_b(&Bg[cur][nl][0], half);
    const v16bf bu = frag_b(&Bu[cur][nl][0], half);
    cg = wmma_bf16(a, bg, cg);
    cu = wmma_bf16(a, bu, cu);
    if (ki + 1 < NK) lstore(cur ^ 1);
    __syncthreads();
  }
  const size_t segbase = (size_t)(e * CAP + mbase + wave * 16);
#pragma unroll
  for (int v = 0; v < 8; ++v) {
    const int M = half * 8 + v;                   // C layout: M = (lane>>4)*8 + vgpr
    hbuf[(segbase + M) * IM + n0 + nl] = (bf16_t)(silu_f(cg[v]) * cu[v]);
  }
}

// ---------------------------------------------------------------------------
// Expert down: out[t] += weight * (h @ Wd^T). Scatter via HW f32 atomics.
// ---------------------------------------------------------------------------
__global__ void k_moe_down(const bf16_t* __restrict__ hbuf, const float* __restrict__ wd,
                           const int* __restrict__ counts, const int* __restrict__ rowmap,
                           const float* __restrict__ wmap, float* __restrict__ out) {
  __shared__ bf16_t As[2][MT][LDP];
  __shared__ bf16_t Bd[2][16][LDP];
  const int e = blockIdx.y >> 3;
  const int mbase = (blockIdx.y & 7) * MT;
  const int ne = counts[e];
  if (mbase >= ne) return;
  const int n0 = blockIdx.x * 16;
  const int tid = threadIdx.x;
  const int lane = tid & 31, wave = tid >> 5;
  const int half = lane >> 4, nl = lane & 15;
  const int arow0 = tid >> 2, aoff = (tid & 3) * 8;
  const int brow = tid >> 4, boff = (tid & 15) * 2;
  const bf16_t* hseg = hbuf + (size_t)(e * CAP + mbase) * IM;
  const bf16_t* pA0 = hseg + (size_t)arow0 * IM + aoff;
  const bf16_t* pA1 = hseg + (size_t)(arow0 + 64) * IM + aoff;
  const float*  pBd = wd + ((size_t)e * H + n0 + brow) * IM + boff;
  v8f c = ZERO8;
  v8bf ra0, ra1; float rd0, rd1;
  auto gload = [&](int k0) {
    ra0 = *(const v8bf*)(pA0 + k0);
    ra1 = *(const v8bf*)(pA1 + k0);
    rd0 = pBd[k0]; rd1 = pBd[k0 + 1];
  };
  auto lstore = [&](int b) {
    *(v8bf*)(&As[b][arow0][aoff]) = ra0;
    *(v8bf*)(&As[b][arow0 + 64][aoff]) = ra1;
    Bd[b][brow][boff] = (bf16_t)rd0; Bd[b][brow][boff + 1] = (bf16_t)rd1;
  };
  gload(0); lstore(0);
  __syncthreads();
  constexpr int NK = IM / KST;
  for (int ki = 0; ki < NK; ++ki) {
    const int cur = ki & 1;
    if (ki + 1 < NK) gload((ki + 1) * KST);
    const v16bf a = frag_a(&As[cur][wave * 16 + nl][0], half);
    const v16bf b = frag_b(&Bd[cur][nl][0], half);
    c = wmma_bf16(a, b, c);
    if (ki + 1 < NK) lstore(cur ^ 1);
    __syncthreads();
  }
#pragma unroll
  for (int v = 0; v < 8; ++v) {
    const int M = half * 8 + v;
    const int r = mbase + wave * 16 + M;
    if (r < ne) {
      const int   tkn = rowmap[e * CAP + r];
      const float w   = wmap[e * CAP + r];
      unsafeAtomicAdd(out + (size_t)tkn * H + n0 + nl, c[v] * w);
    }
  }
}

// ---------------------------------------------------------------------------
// Shared expert gate/up (dense over all tokens) -> sbuf (bf16).
// ---------------------------------------------------------------------------
__global__ void k_sh_gateup(const bf16_t* __restrict__ xbf, const float* __restrict__ wsg,
                            const float* __restrict__ wsu, bf16_t* __restrict__ sbuf) {
  __shared__ bf16_t As[2][MT][LDP];
  __shared__ bf16_t Bg[2][16][LDP];
  __shared__ bf16_t Bu[2][16][LDP];
  const int mbase = blockIdx.y * MT;
  const int n0 = blockIdx.x * 16;
  const int tid = threadIdx.x;
  const int lane = tid & 31, wave = tid >> 5;
  const int half = lane >> 4, nl = lane & 15;
  const int arow0 = tid >> 2, aoff = (tid & 3) * 8;
  const int brow = tid >> 4, boff = (tid & 15) * 2;
  const bf16_t* pA0 = xbf + (size_t)(mbase + arow0) * H + aoff;
  const bf16_t* pA1 = xbf + (size_t)(mbase + arow0 + 64) * H + aoff;
  const float*  pBg = wsg + (size_t)(n0 + brow) * H + boff;
  const float*  pBu = wsu + (size_t)(n0 + brow) * H + boff;
  v8f cg = ZERO8, cu = ZERO8;
  v8bf ra0, ra1; float rg0, rg1, ru0, ru1;
  auto gload = [&](int k0) {
    ra0 = *(const v8bf*)(pA0 + k0);
    ra1 = *(const v8bf*)(pA1 + k0);
    rg0 = pBg[k0]; rg1 = pBg[k0 + 1];
    ru0 = pBu[k0]; ru1 = pBu[k0 + 1];
  };
  auto lstore = [&](int b) {
    *(v8bf*)(&As[b][arow0][aoff]) = ra0;
    *(v8bf*)(&As[b][arow0 + 64][aoff]) = ra1;
    Bg[b][brow][boff] = (bf16_t)rg0; Bg[b][brow][boff + 1] = (bf16_t)rg1;
    Bu[b][brow][boff] = (bf16_t)ru0; Bu[b][brow][boff + 1] = (bf16_t)ru1;
  };
  gload(0); lstore(0);
  __syncthreads();
  constexpr int NK = H / KST;
  for (int ki = 0; ki < NK; ++ki) {
    const int cur = ki & 1;
    if (ki + 1 < NK) gload((ki + 1) * KST);
    const v16bf a  = frag_a(&As[cur][wave * 16 + nl][0], half);
    const v16bf bg = frag_b(&Bg[cur][nl][0], half);
    const v16bf bu = frag_b(&Bu[cur][nl][0], half);
    cg = wmma_bf16(a, bg, cg);
    cu = wmma_bf16(a, bu, cu);
    if (ki + 1 < NK) lstore(cur ^ 1);
    __syncthreads();
  }
#pragma unroll
  for (int v = 0; v < 8; ++v) {
    const int M = half * 8 + v;
    sbuf[(size_t)(mbase + wave * 16 + M) * ISH + n0 + nl] =
        (bf16_t)(silu_f(cg[v]) * cu[v]);
  }
}

// ---------------------------------------------------------------------------
// Shared expert down: out[t][n] = sigmoid(gate[t]) * (s @ Wsd^T).
// Plain store (single writer) -> initializes out before expert-down atomics.
// ---------------------------------------------------------------------------
__global__ void k_sh_down(const bf16_t* __restrict__ sbuf, const float* __restrict__ wsd,
                          const float* __restrict__ sgate, float* __restrict__ out) {
  __shared__ bf16_t As[2][MT][LDP];
  __shared__ bf16_t Bd[2][16][LDP];
  const int mbase = blockIdx.y * MT;
  const int n0 = blockIdx.x * 16;
  const int tid = threadIdx.x;
  const int lane = tid & 31, wave = tid >> 5;
  const int half = lane >> 4, nl = lane & 15;
  const int arow0 = tid >> 2, aoff = (tid & 3) * 8;
  const int brow = tid >> 4, boff = (tid & 15) * 2;
  const bf16_t* pA0 = sbuf + (size_t)(mbase + arow0) * ISH + aoff;
  const bf16_t* pA1 = sbuf + (size_t)(mbase + arow0 + 64) * ISH + aoff;
  const float*  pBd = wsd + (size_t)(n0 + brow) * ISH + boff;
  v8f c = ZERO8;
  v8bf ra0, ra1; float rd0, rd1;
  auto gload = [&](int k0) {
    ra0 = *(const v8bf*)(pA0 + k0);
    ra1 = *(const v8bf*)(pA1 + k0);
    rd0 = pBd[k0]; rd1 = pBd[k0 + 1];
  };
  auto lstore = [&](int b) {
    *(v8bf*)(&As[b][arow0][aoff]) = ra0;
    *(v8bf*)(&As[b][arow0 + 64][aoff]) = ra1;
    Bd[b][brow][boff] = (bf16_t)rd0; Bd[b][brow][boff + 1] = (bf16_t)rd1;
  };
  gload(0); lstore(0);
  __syncthreads();
  constexpr int NK = ISH / KST;
  for (int ki = 0; ki < NK; ++ki) {
    const int cur = ki & 1;
    if (ki + 1 < NK) gload((ki + 1) * KST);
    const v16bf a = frag_a(&As[cur][wave * 16 + nl][0], half);
    const v16bf b = frag_b(&Bd[cur][nl][0], half);
    c = wmma_bf16(a, b, c);
    if (ki + 1 < NK) lstore(cur ^ 1);
    __syncthreads();
  }
#pragma unroll
  for (int v = 0; v < 8; ++v) {
    const int M = half * 8 + v;
    const int t = mbase + wave * 16 + M;
    out[(size_t)t * H + n0 + nl] = sigmoid_f(sgate[t]) * c[v];
  }
}

// ---------------------------------------------------------------------------
extern "C" void kernel_launch(void* const* d_in, const int* in_sizes, int n_in,
                              void* d_out, int out_size, void* d_ws, size_t ws_size,
                              hipStream_t stream) {
  (void)in_sizes; (void)n_in; (void)out_size; (void)ws_size;
  const float* x    = (const float*)d_in[0];
  const float* wr   = (const float*)d_in[1];
  const float* wg   = (const float*)d_in[2];
  const float* wu   = (const float*)d_in[3];
  const float* wd   = (const float*)d_in[4];
  const float* wsg  = (const float*)d_in[5];
  const float* wsu  = (const float*)d_in[6];
  const float* wsd  = (const float*)d_in[7];
  const float* wshg = (const float*)d_in[8];
  float* out    = (float*)d_out;
  float* logits = out + (size_t)T * H;   // outputs concatenated: final, router_logits

  // workspace carve-out (~62 MB)
  char* ws = (char*)d_ws;
  size_t off = 0;
  auto carve = [&](size_t bytes) -> void* {
    void* p = ws + off;
    off = (off + bytes + 255) & ~(size_t)255;
    return p;
  };
  int*    counts = (int*)carve(E * sizeof(int));
  int*    rowmap = (int*)carve((size_t)E * CAP * sizeof(int));
  float*  wmap   = (float*)carve((size_t)E * CAP * sizeof(float));
  float*  sgate  = (float*)carve((size_t)T * sizeof(float));
  bf16_t* xbf    = (bf16_t*)carve((size_t)T * H * sizeof(bf16_t));
  bf16_t* hbuf   = (bf16_t*)carve((size_t)E * CAP * IM * sizeof(bf16_t));
  bf16_t* sbuf   = (bf16_t*)carve((size_t)T * ISH * sizeof(bf16_t));

  k_init  <<<1, 32, 0, stream>>>(counts);
  k_router<<<T, 256, 0, stream>>>(x, wr, wshg, logits, sgate);
  k_topk  <<<(T + 255) / 256, 256, 0, stream>>>(logits, counts, rowmap, wmap);
  k_cvtx  <<<(T * H) / 1024, 256, 0, stream>>>(x, xbf);

  k_sh_gateup <<<dim3(ISH / 16, T / MT), 256, 0, stream>>>(xbf, wsg, wsu, sbuf);
  k_sh_down   <<<dim3(H / 16,  T / MT), 256, 0, stream>>>(sbuf, wsd, sgate, out);

  k_moe_gateup<<<dim3(IM / 16, E * (CAP / MT)), 256, 0, stream>>>(xbf, wg, wu, counts, rowmap, hbuf);
  k_moe_down  <<<dim3(H / 16,  E * (CAP / MT)), 256, 0, stream>>>(hbuf, wd, counts, rowmap, wmap, out);
}
